// EfficientFusionLayer_34995393528522
// MI455X (gfx1250) — matrix operand order repaired and checked by reading, not compiled, not run on hardware
//
#include <hip/hip_runtime.h>

// ---------------------------------------------------------------------------
// EfficientFusionLayer on gfx1250 (MI455X): bf16 WMMA GEMMs with algebraic
// fusion attn1(x) = x @ (Wo@Wv)^T + (Wo@bv + bo).
// B=16384, DIM=1024, HID=2048. All heavy math -> v_wmma_f32_16x16x32_bf16.
// GEMM: 8 wave32 per WG, wave tile 64x64 (16 WMMA accums), WG tile 128x256.
// K is a template constant so all 8 fragment addresses fold into immediate
// offsets of a single base pointer (no per-iteration 64-bit address VALU).
// ---------------------------------------------------------------------------

typedef __attribute__((ext_vector_type(16))) __bf16 v16bf;
typedef __attribute__((ext_vector_type(8)))  float  v8f;
typedef unsigned short u16;

#define Bn   16384
#define DIMn 1024
#define HIDn 2048

__device__ __forceinline__ u16 f2bf(float f) {
    union { float f; unsigned u; } cv; cv.f = f;
    unsigned u = cv.u;
    u += 0x7FFFu + ((u >> 16) & 1u);   // round-to-nearest-even
    return (u16)(u >> 16);
}

// --------------------------- operand packing -------------------------------
// A-operand layout (16x32 bf16 per tile, ISA 7.12.2):
// for k: g=(k%32)>>3 ; lane = (m&15) + 16*(g&1) ; slot = (k&7) | ((g>>1)<<3).
__global__ void pack_a_f32(const float* __restrict__ src, u16* __restrict__ dst,
                           int M, int K) {
    int idx = blockIdx.x * blockDim.x + threadIdx.x;   // one 8-elem octet each
    int KG = K >> 3;
    if (idx >= M * KG) return;
    int m = idx / KG, kg = idx % KG, k0 = kg << 3;
    int kc = k0 & 31, g = kc >> 3;
    int l = (m & 15) + ((g & 1) << 4);
    size_t base = ((size_t)(m >> 4) * (K >> 5) + (k0 >> 5)) * 512
                + (size_t)l * 16 + ((g >> 1) << 3);
    const float* s = src + (size_t)m * K + k0;
    float4 f0 = *(const float4*)s, f1 = *(const float4*)(s + 4);
    union { u16 h[8]; uint4 v; } pk;
    pk.h[0]=f2bf(f0.x); pk.h[1]=f2bf(f0.y); pk.h[2]=f2bf(f0.z); pk.h[3]=f2bf(f0.w);
    pk.h[4]=f2bf(f1.x); pk.h[5]=f2bf(f1.y); pk.h[6]=f2bf(f1.z); pk.h[7]=f2bf(f1.w);
    *(uint4*)(dst + base) = pk.v;
}

// B-operand layout (32x16): lane l holds column n=l&15, K-halves by l>>4,
// slot s holds k=8h+s (s<8) or 16+8h+(s-8) (s>=8), h=l>>4, same as A.
template <bool TRANS>
__global__ void pack_b_f32(const float* __restrict__ src, u16* __restrict__ dst,
                           int N, int K, int srcLD) {
    int idx = blockIdx.x * blockDim.x + threadIdx.x;
    int KG = K >> 3;
    if (idx >= N * KG) return;
    int n = idx / KG, kg = idx % KG, k0 = kg << 3;
    int kc = k0 & 31, g = kc >> 3;
    int l = (n & 15) + ((g & 1) << 4);
    size_t base = ((size_t)(n >> 4) * (K >> 5) + (k0 >> 5)) * 512
                + (size_t)l * 16 + ((g >> 1) << 3);
    union { u16 h[8]; uint4 v; } pk;
    if (TRANS) {    // logical W[n,k] = src[k*srcLD + n]
        #pragma unroll
        for (int e = 0; e < 8; ++e)
            pk.h[e] = f2bf(src[(size_t)(k0 + e) * srcLD + n]);
    } else {
        const float* s = src + (size_t)n * K + k0;
        float4 f0 = *(const float4*)s, f1 = *(const float4*)(s + 4);
        pk.h[0]=f2bf(f0.x); pk.h[1]=f2bf(f0.y); pk.h[2]=f2bf(f0.z); pk.h[3]=f2bf(f0.w);
        pk.h[4]=f2bf(f1.x); pk.h[5]=f2bf(f1.y); pk.h[6]=f2bf(f1.z); pk.h[7]=f2bf(f1.w);
    }
    *(uint4*)(dst + base) = pk.v;
}

// ------------------------------- GEMM --------------------------------------
// C[M,N] = A[M,K] @ W[N,K]^T, bf16 operands pre-packed, f32 accumulate.
// 256 thr = 8 wave32 arranged 2(M) x 4(N); wave tile 64x64 = 16 WMMA accums.
// KT = K/32 (compile-time) so fragment strides become immediate offsets.
// MODE 0: C -> outF                     (used for Wc = Wo@Wv)
// MODE 1: C + bias[n] + res0 -> outF    (self-attn pre-LN; FFN2 + residual)
// MODE 2: res0 + 0.5*(res1 + C + bias)  (fused-average pre-LN)
// MODE 3: relu(C + bias) -> outP packed bf16 (FFN1 feeding FFN2 directly)
template <int MODE, int KT>
__global__ __launch_bounds__(256) void gemm_wmma(
    const u16* __restrict__ Ap, const u16* __restrict__ Bp,
    int M, int N,
    const float* __restrict__ bias, const float* __restrict__ res0,
    const float* __restrict__ res1, float* __restrict__ outF,
    u16* __restrict__ outP) {
    const int lane = threadIdx.x & 31;
    const int w    = threadIdx.x >> 5;
    const int wm   = w & 1, wn = w >> 1;
    const size_t mt0 = (size_t)blockIdx.y * 8  + wm * 4;   // 4 m-tiles
    const size_t nt0 = (size_t)blockIdx.x * 16 + wn * 4;   // 4 n-tiles
    const v16bf* Ab = (const v16bf*)Ap + mt0 * KT * 32 + lane;
    const v16bf* Bb = (const v16bf*)Bp + nt0 * KT * 32 + lane;

    v8f acc[4][4];
    const v8f vzero = {0.f,0.f,0.f,0.f,0.f,0.f,0.f,0.f};
    #pragma unroll
    for (int i = 0; i < 4; ++i)
        #pragma unroll
        for (int j = 0; j < 4; ++j) acc[i][j] = vzero;

    #pragma unroll 2
    for (int kt = 0; kt < KT; ++kt) {
        v16bf a[4], b[4];
        #pragma unroll
        for (int i = 0; i < 4; ++i)
            a[i] = Ab[(size_t)i * KT * 32];     // constant byte offsets
        #pragma unroll
        for (int j = 0; j < 4; ++j)
            b[j] = Bb[(size_t)j * KT * 32];
        Ab += 32;                               // single bump per operand
        Bb += 32;
        #pragma unroll
        for (int i = 0; i < 4; ++i)
            #pragma unroll
            for (int j = 0; j < 4; ++j)
                acc[i][j] = __builtin_amdgcn_wmma_f32_16x16x32_bf16(
                    false, a[i], false, b[j], (short)0, acc[i][j], false, false);
    }

    // C/D layout: vgpr v, lane l -> m_local = v + 8*(l>>4), n_local = l&15
    const int mbase = blockIdx.y * 128 + wm * 64 + (lane >> 4) * 8;
    const int nbase = blockIdx.x * 256 + wn * 64 + (lane & 15);
    #pragma unroll
    for (int i = 0; i < 4; ++i) {
        #pragma unroll
        for (int j = 0; j < 4; ++j) {
            #pragma unroll
            for (int v = 0; v < 8; ++v) {
                int row = mbase + i * 16 + v;
                int col = nbase + j * 16;
                float a = acc[i][j][v];
                size_t idx = (size_t)row * N + col;
                if (MODE == 0) {
                    outF[idx] = a;
                } else if (MODE == 1) {
                    outF[idx] = a + bias[col] + res0[idx];
                } else if (MODE == 2) {
                    outF[idx] = res0[idx] + 0.5f * (res1[idx] + a + bias[col]);
                } else { // MODE 3: relu + direct A-operand pack for next GEMM
                    a += bias[col];
                    a = a > 0.f ? a : 0.f;
                    int kc = col & 31, g = kc >> 3;
                    int l2 = (row & 15) + ((g & 1) << 4);
                    int s2 = (kc & 7) | ((g >> 1) << 3);
                    size_t d = ((size_t)(row >> 4) * (N >> 5) + (col >> 5)) * 512
                             + (size_t)l2 * 16 + s2;
                    outP[d] = f2bf(a);
                }
            }
        }
    }
}

// ------------------------------ LayerNorm ----------------------------------
// One row (K=1024) per 256-thread block; emits f32 row + packed-bf16 A-operand.
__global__ __launch_bounds__(256) void ln_pack(
    const float* __restrict__ in, const float* __restrict__ gam,
    const float* __restrict__ bet, float* __restrict__ outF,
    u16* __restrict__ outP, int K) {
    int row = blockIdx.x, tid = threadIdx.x;
    int lane = tid & 31, w = tid >> 5;
    int k0 = tid * 4;
    float4 x = *(const float4*)(in + (size_t)row * K + k0);
    float s  = x.x + x.y + x.z + x.w;
    float ss = x.x*x.x + x.y*x.y + x.z*x.z + x.w*x.w;
    #pragma unroll
    for (int off = 16; off >= 1; off >>= 1) {
        s  += __shfl_xor(s,  off, 32);
        ss += __shfl_xor(ss, off, 32);
    }
    __shared__ float r1[8], r2[8];
    if (lane == 0) { r1[w] = s; r2[w] = ss; }
    __syncthreads();
    float tot = 0.f, tots = 0.f;
    #pragma unroll
    for (int i = 0; i < 8; ++i) { tot += r1[i]; tots += r2[i]; }
    float mean = tot * (1.0f / K);
    float var  = tots * (1.0f / K) - mean * mean;
    float rinv = rsqrtf(var + 1e-5f);
    float4 gg = *(const float4*)(gam + k0);
    float4 bb = *(const float4*)(bet + k0);
    float y0 = (x.x - mean) * rinv * gg.x + bb.x;
    float y1 = (x.y - mean) * rinv * gg.y + bb.y;
    float y2 = (x.z - mean) * rinv * gg.z + bb.z;
    float y3 = (x.w - mean) * rinv * gg.w + bb.w;
    *(float4*)(outF + (size_t)row * K + k0) = make_float4(y0, y1, y2, y3);
    int kc = k0 & 31, g = kc >> 3;
    int l  = (row & 15) + ((g & 1) << 4);
    int s2 = (kc & 7) | ((g >> 1) << 3);
    size_t d = ((size_t)(row >> 4) * (K >> 5) + (k0 >> 5)) * 512
             + (size_t)l * 16 + s2;
    union { u16 h[4]; uint2 v; } pk;
    pk.h[0]=f2bf(y0); pk.h[1]=f2bf(y1); pk.h[2]=f2bf(y2); pk.h[3]=f2bf(y3);
    *(uint2*)(outP + d) = pk.v;
}

// bc[n] = Wo[n,:] . bv + bo[n]
__global__ void combine_bias(const float* __restrict__ Wo,
                             const float* __restrict__ bv,
                             const float* __restrict__ bo,
                             float* __restrict__ bc, int D) {
    int n = blockIdx.x * blockDim.x + threadIdx.x;
    if (n >= D) return;
    float a = 0.f;
    for (int m = 0; m < D; ++m) a += Wo[(size_t)n * D + m] * bv[m];
    bc[n] = a + bo[n];
}

__global__ void avg_kernel(const float* __restrict__ a,
                           const float* __restrict__ b,
                           float* __restrict__ o, int n4) {
    int i = blockIdx.x * blockDim.x + threadIdx.x;
    if (i >= n4) return;
    float4 x = ((const float4*)a)[i];
    float4 y = ((const float4*)b)[i];
    ((float4*)o)[i] = make_float4(0.5f*(x.x+y.x), 0.5f*(x.y+y.y),
                                  0.5f*(x.z+y.z), 0.5f*(x.w+y.w));
}

// ------------------------------- launcher ----------------------------------
extern "C" void kernel_launch(void* const* d_in, const int* in_sizes, int n_in,
                              void* d_out, int out_size, void* d_ws, size_t ws_size,
                              hipStream_t stream) {
    const size_t MBy = 1024ull * 1024ull;
    const size_t BD  = (size_t)Bn * DIMn;
    char* ws = (char*)d_ws;
    // workspace map (~481 MB total)
    u16*   WCP   = (u16*)(ws + 0);          // 4 x 1024x1024 bf16 packed (8MB)
    u16*   W1P   = (u16*)(ws + 8  * MBy);   // 2 x 2048x1024 (8MB)
    u16*   W2P   = (u16*)(ws + 16 * MBy);   // 2 x 1024x2048 (8MB)
    float* BC    = (float*)(ws + 24 * MBy); // 4 x 1024 f32
    float* TWC   = (float*)(ws + 25 * MBy); // Wc f32 scratch (4MB)
    u16*   TA    = (u16*)(ws + 29 * MBy);   // packed Wo (2MB)
    u16*   TB    = (u16*)(ws + 31 * MBy);   // packed Wv^T (2MB)
    u16*   P0    = (u16*)(ws + 33 * MBy);   // packed act v (32MB)
    u16*   P1    = (u16*)(ws + 65 * MBy);   // packed act t (32MB)
    u16*   PH0   = (u16*)(ws + 97 * MBy);   // packed hidden v (64MB)
    u16*   PH1   = (u16*)(ws + 161 * MBy);  // packed hidden t (64MB)
    float* F0    = (float*)(ws + 225 * MBy); // pre-LN v (64MB)
    float* F1    = (float*)(ws + 289 * MBy); // pre-LN t
    float* F2    = (float*)(ws + 353 * MBy); // v_self -> v_fused f32
    float* F3    = (float*)(ws + 417 * MBy); // t_self -> t_fused f32

    const float* vis = (const float*)d_in[0];
    const float* txt = (const float*)d_in[1];
    float* out = (float*)d_out;

    const int PKG_W  = (DIMn * DIMn / 8 + 255) / 256;   // 1024x1024 octets
    const int PKG_W1 = (HIDn * DIMn / 8 + 255) / 256;
    const int PKG_X  = ((int)(BD / 8) + 255) / 256;

    // ---- weight prep: Wc = Wo @ Wv (tiny WMMA GEMM), bc = Wo@bv + bo ----
    for (int t = 0; t < 4; ++t) {
        const float* Wv = (const float*)d_in[2 + 4 * t];
        const float* bv = (const float*)d_in[3 + 4 * t];
        const float* Wo = (const float*)d_in[4 + 4 * t];
        const float* bo = (const float*)d_in[5 + 4 * t];
        pack_a_f32<<<PKG_W, 256, 0, stream>>>(Wo, TA, DIMn, DIMn);
        pack_b_f32<true><<<PKG_W, 256, 0, stream>>>(Wv, TB, DIMn, DIMn, DIMn);
        gemm_wmma<0, DIMn / 32><<<dim3(4, 8), 256, 0, stream>>>(
            TA, TB, DIMn, DIMn, nullptr, nullptr, nullptr, TWC, nullptr);
        pack_b_f32<false><<<PKG_W, 256, 0, stream>>>(
            TWC, WCP + (size_t)t * DIMn * DIMn, DIMn, DIMn, DIMn);
        combine_bias<<<4, 256, 0, stream>>>(Wo, bv, bo, BC + t * DIMn, DIMn);
    }
    pack_b_f32<false><<<PKG_W1, 256, 0, stream>>>((const float*)d_in[26], W1P,                      HIDn, DIMn, DIMn);
    pack_b_f32<false><<<PKG_W1, 256, 0, stream>>>((const float*)d_in[30], W1P + (size_t)HIDn*DIMn, HIDn, DIMn, DIMn);
    pack_b_f32<false><<<PKG_W1, 256, 0, stream>>>((const float*)d_in[28], W2P,                      DIMn, HIDn, HIDn);
    pack_b_f32<false><<<PKG_W1, 256, 0, stream>>>((const float*)d_in[32], W2P + (size_t)DIMn*HIDn, DIMn, HIDn, HIDn);

    // ---- stage 1: self-attn (+residual) -> LN1 ----
    pack_a_f32<<<PKG_X, 256, 0, stream>>>(vis, P0, Bn, DIMn);
    pack_a_f32<<<PKG_X, 256, 0, stream>>>(txt, P1, Bn, DIMn);
    dim3 gD(DIMn / 256, Bn / 128), gH(HIDn / 256, Bn / 128);
    gemm_wmma<1, DIMn / 32><<<gD, 256, 0, stream>>>(P0, WCP + 0ull*DIMn*DIMn, Bn, DIMn,
                                         BC + 0*DIMn, vis, nullptr, F0, nullptr);
    gemm_wmma<1, DIMn / 32><<<gD, 256, 0, stream>>>(P1, WCP + 1ull*DIMn*DIMn, Bn, DIMn,
                                         BC + 1*DIMn, txt, nullptr, F1, nullptr);
    ln_pack<<<Bn, 256, 0, stream>>>(F0, (const float*)d_in[18], (const float*)d_in[19], F2, P0, DIMn);
    ln_pack<<<Bn, 256, 0, stream>>>(F1, (const float*)d_in[20], (const float*)d_in[21], F3, P1, DIMn);

    // ---- stage 2: cross-attn, fused average (+residual) -> LN2 ----
    gemm_wmma<2, DIMn / 32><<<gD, 256, 0, stream>>>(P1, WCP + 2ull*DIMn*DIMn, Bn, DIMn,
                                         BC + 2*DIMn, vis, F2, F0, nullptr); // v_cross from t_self
    gemm_wmma<2, DIMn / 32><<<gD, 256, 0, stream>>>(P0, WCP + 3ull*DIMn*DIMn, Bn, DIMn,
                                         BC + 3*DIMn, txt, F3, F1, nullptr); // t_cross from v_self
    ln_pack<<<Bn, 256, 0, stream>>>(F0, (const float*)d_in[22], (const float*)d_in[23], F2, P0, DIMn);
    ln_pack<<<Bn, 256, 0, stream>>>(F1, (const float*)d_in[24], (const float*)d_in[25], F3, P1, DIMn);

    // ---- stage 3: FFN1 (relu, writes packed bf16 hidden directly) ----
    gemm_wmma<3, DIMn / 32><<<gH, 256, 0, stream>>>(P0, W1P,                      Bn, HIDn,
                                         (const float*)d_in[27], nullptr, nullptr, nullptr, PH0);
    gemm_wmma<3, DIMn / 32><<<gH, 256, 0, stream>>>(P1, W1P + (size_t)HIDn*DIMn, Bn, HIDn,
                                         (const float*)d_in[31], nullptr, nullptr, nullptr, PH1);

    // ---- stage 4: FFN2 (+fused residual) straight into d_out ----
    gemm_wmma<1, HIDn / 32><<<gD, 256, 0, stream>>>(PH0, W2P,                      Bn, DIMn,
                                         (const float*)d_in[29], F2, nullptr, out, nullptr);
    gemm_wmma<1, HIDn / 32><<<gD, 256, 0, stream>>>(PH1, W2P + (size_t)DIMn*HIDn, Bn, DIMn,
                                         (const float*)d_in[33], F3, nullptr, out + BD, nullptr);

    // ---- new_context = 0.5*(v_out + t_out) ----
    avg_kernel<<<(int)(BD / 4 + 255) / 256, 256, 0, stream>>>(out, out + BD, out + 2 * BD, (int)(BD / 4));
}
